// FilterGNN_57904749084726
// MI455X (gfx1250) — matrix operand order repaired and checked by reading, not compiled, not run on hardware
//
#include <hip/hip_runtime.h>
#include <math.h>

// ---------------------------------------------------------------------------
// FilterGNN on MI455X (gfx1250, wave32).
// Dense lin_l / bias GEMMs -> v_wmma_f32_16x16x32_f16, one wave per 16x(16*NT)
// output strip (A fragment reused across NT WMMAs -> 4x less A traffic).
// Graph ops (mean aggregation, cosine attention, scatter-add) stay fp32 and
// L2-resident (each node tensor ~50MB << 192MB L2); gathers use b128 loads.
// ---------------------------------------------------------------------------

typedef __attribute__((ext_vector_type(16))) _Float16 v16h;
typedef __attribute__((ext_vector_type(8)))  _Float16 v8h;
typedef __attribute__((ext_vector_type(8)))  float    v8f;

#define LN_EPSF  1e-5f
#define COS_EPSF 1e-8f

static constexpr int N_NODES = 50000;
static constexpr int N_EDGES = 500000;

// -------------------------- small device helpers ---------------------------

__device__ __forceinline__ float wave_sum32(float v) {
#pragma unroll
  for (int o = 16; o > 0; o >>= 1) v += __shfl_xor(v, o, 32);
  return v;
}
__device__ __forceinline__ float wave_max32(float v) {
#pragma unroll
  for (int o = 16; o > 0; o >>= 1) v = fmaxf(v, __shfl_xor(v, o, 32));
  return v;
}

// ------------------------------- utility -----------------------------------

__global__ void k_zero(float* p, long n) {
  long i = (long)blockIdx.x * blockDim.x + threadIdx.x;
  if (i < n) p[i] = 0.0f;
}

__global__ void k_cvt_f16(_Float16* o, const float* i, long n) {
  long t = (long)blockIdx.x * blockDim.x + threadIdx.x;
  if (t < n) o[t] = (_Float16)i[t];
}

// W (K x Nc, row-major f32) -> Wt (Npad x K, row-major f16), zero-padded rows.
__global__ void k_wtrans(_Float16* o, const float* W, int K, int Nc, int Npad) {
  long t = (long)blockIdx.x * blockDim.x + threadIdx.x;
  long total = (long)Npad * K;
  if (t >= total) return;
  int r = (int)(t / K);   // output row == weight column
  int c = (int)(t % K);   // K index
  o[t] = (r < Nc) ? (_Float16)W[(long)c * Nc + r] : (_Float16)0.0f;
}

// ------------------------------ graph prep ---------------------------------

__global__ void k_deg(float* deg, const int* src, int E) {
  int e = blockIdx.x * blockDim.x + threadIdx.x;
  if (e < E) atomicAdd(&deg[src[e]], 1.0f);
}

// mu[src] += x[dst]  (4 channels / thread, b128 gather)
__global__ void k_mu_acc(float* mu, const float* x, const int* src, const int* dst,
                         int E, int C) {
  long t = (long)blockIdx.x * blockDim.x + threadIdx.x;
  int c4n = C >> 2;
  long total = (long)E * c4n;
  if (t >= total) return;
  int e  = (int)(t / c4n);
  int c4 = (int)(t % c4n);
  int s = src[e], d = dst[e];
  float4 v = ((const float4*)(x + (long)d * C))[c4];
  float* mp = mu + (long)s * C + (c4 << 2);
  atomicAdd(&mp[0], v.x);
  atomicAdd(&mp[1], v.y);
  atomicAdd(&mp[2], v.z);
  atomicAdd(&mp[3], v.w);
}

__global__ void k_mu_div(float* mu, const float* deg, int N, int C) {
  long t = (long)blockIdx.x * blockDim.x + threadIdx.x;
  if (t >= (long)N * C) return;
  int n = (int)(t / C);
  mu[t] /= fmaxf(deg[n], 1.0f);
}

// L2 norm per node row of mu (wave per node, b128 loads)
__global__ void k_norm(float* nrm, const float* mu, int N, int C) {
  int wid  = (blockIdx.x * blockDim.x + threadIdx.x) >> 5;
  int lane = threadIdx.x & 31;
  if (wid >= N) return;
  const float4* row = (const float4*)(mu + (long)wid * C);
  float s = 0.0f;
  for (int i = lane; i < (C >> 2); i += 32) {
    float4 v = row[i];
    s += v.x * v.x + v.y * v.y + v.z * v.z + v.w * v.w;
  }
  s = wave_sum32(s);
  if (lane == 0) nrm[wid] = sqrtf(s);
}

// ------------------------------ WMMA GEMM ----------------------------------
// out[M x Nc] = A16[M x K] * Wt16^T   (Wt16 stored as Npad x K: B columns
// contiguous along K).  One wave computes a 16 x (16*NT) strip: the A fragment
// is loaded once per K-step and reused by NT WMMAs (independent acc chains).
// Register layouts per CDNA5 ISA 7.12.2:
//   A (16x32 f16): lane g=lane>>4, m=lane&15; halves 0..7 = A[m][g*8+0..7],
//                  halves 8..15 = A[m][16+g*8+0..7]
//   B (32x16 f16): same k order as A: halves 0..7 = Wt[col m][kk+g*8+0..7],
//                  halves 8..15 = Wt[col m][kk+16+g*8+0..7]
//   D (16x16 f32): vgpr r -> row r+8*g, col m
template <int NT>
__global__ void k_gemm_wmma(const _Float16* __restrict__ A,
                            const _Float16* __restrict__ Bt,
                            const float* __restrict__ bias,
                            float* __restrict__ out,
                            int M, int K, int Npad, int Nc, int ldo) {
  int wave = (blockIdx.x * blockDim.x + threadIdx.x) >> 5;
  int lane = threadIdx.x & 31;
  int nstrips = (Npad >> 4) / NT;
  int mtiles  = M >> 4;          // M multiple of 16 (50000 = 3125*16)
  int tm  = wave / nstrips;
  int tn0 = (wave % nstrips) * NT;
  if (tm >= mtiles) return;      // whole-wave uniform exit; EXEC stays full

  int g = lane >> 4;             // lane group (0: lanes 0-15, 1: lanes 16-31)
  int m = lane & 15;

  const _Float16* arow = A + (long)(tm * 16 + m) * K;
  const _Float16* bcol[NT];
#pragma unroll
  for (int t = 0; t < NT; ++t)
    bcol[t] = Bt + (long)((tn0 + t) * 16 + m) * K;

  v8f acc[NT];
#pragma unroll
  for (int t = 0; t < NT; ++t) acc[t] = {};

  for (int kk = 0; kk < K; kk += 32) {
    v8h alo = *(const v8h*)(arow + kk + g * 8);
    v8h ahi = *(const v8h*)(arow + kk + 16 + g * 8);
    v16h a;
#pragma unroll
    for (int i = 0; i < 8; ++i) { a[i] = alo[i]; a[8 + i] = ahi[i]; }
#pragma unroll
    for (int t = 0; t < NT; ++t) {
      v8h blo = *(const v8h*)(bcol[t] + kk + g * 8);
      v8h bhi = *(const v8h*)(bcol[t] + kk + 16 + g * 8);
      v16h b;
#pragma unroll
      for (int i = 0; i < 8; ++i) { b[i] = blo[i]; b[8 + i] = bhi[i]; }
      acc[t] = __builtin_amdgcn_wmma_f32_16x16x32_f16(
          /*neg_a=*/false, a, /*neg_b=*/false, b,
          /*c_mod=*/(short)0, acc[t], /*reuse_a=*/false, /*reuse_b=*/false);
    }
  }

#pragma unroll
  for (int t = 0; t < NT; ++t) {
    int col = (tn0 + t) * 16 + m;
    if (col < Nc) {
      float bv = bias ? bias[col] : 0.0f;
#pragma unroll
      for (int r = 0; r < 8; ++r) {
        int row = tm * 16 + r + g * 8;
        out[(long)row * ldo + col] = acc[t][r] + bv;
      }
    }
  }
}

// ------------------------------- LayerNorm ---------------------------------
// In-place LN over each (node, head) block of Ch channels; wave per block.
__global__ void k_ln(float* x, const float* __restrict__ gam,
                     const float* __restrict__ bet, int NH, int Ch) {
  int wid  = (blockIdx.x * blockDim.x + threadIdx.x) >> 5;
  int lane = threadIdx.x & 31;
  if (wid >= NH) return;
  float* row = x + (long)wid * Ch;
  float s = 0.0f, s2 = 0.0f;
  for (int c = lane; c < Ch; c += 32) {
    float v = row[c];
    s += v; s2 += v * v;
  }
  s  = wave_sum32(s);
  s2 = wave_sum32(s2);
  float mean = s / (float)Ch;
  float var  = s2 / (float)Ch - mean * mean;
  float rs   = rsqrtf(var + LN_EPSF);
  for (int c = lane; c < Ch; c += 32)
    row[c] = (row[c] - mean) * rs * gam[c] + bet[c];
}

// ----------------------- attention message passing -------------------------

// self loop: alpha = ||mu||^2 / max(||mu||^2, eps); agg[n] += xl[n]*alpha
__global__ void k_self(float* agg, const float* __restrict__ xl,
                       const float* __restrict__ nrm, int N, int C) {
  long t = (long)blockIdx.x * blockDim.x + threadIdx.x;
  if (t >= (long)N * C) return;
  int n = (int)(t / C);
  float nn = nrm[n] * nrm[n];
  float alpha = nn / fmaxf(nn, COS_EPSF);
  agg[t] += xl[t] * alpha;
}

// wave per edge: cosine(mu[s], mu[d]) then scatter xl[s]*alpha into agg[d].
// Vector (b128) paths when the channel count is a multiple of 128.
__global__ void k_edge(float* agg, const float* __restrict__ xl,
                       const float* __restrict__ mu,
                       const float* __restrict__ nrm,
                       const int* __restrict__ src, const int* __restrict__ dst,
                       int E, int Cin, int Cout) {
  int wid  = (blockIdx.x * blockDim.x + threadIdx.x) >> 5;
  int lane = threadIdx.x & 31;
  if (wid >= E) return;
  int s = src[wid], d = dst[wid];
  const float* ms = mu + (long)s * Cin;
  const float* md = mu + (long)d * Cin;

  float dot = 0.0f;
  if ((Cin & 127) == 0) {
    const float4* ms4 = (const float4*)ms;
    const float4* md4 = (const float4*)md;
    for (int i = lane; i < (Cin >> 2); i += 32) {
      float4 a = ms4[i], b = md4[i];
      dot += a.x * b.x + a.y * b.y + a.z * b.z + a.w * b.w;
    }
  } else {
    for (int c = lane; c < Cin; c += 32) dot += ms[c] * md[c];
  }
  dot = wave_sum32(dot);
  float alpha = dot / fmaxf(nrm[s] * nrm[d], COS_EPSF);

  const float* xs = xl + (long)s * Cout;
  float* ad = agg + (long)d * Cout;
  if ((Cout & 127) == 0) {
    const float4* xs4 = (const float4*)xs;
    for (int i = lane; i < (Cout >> 2); i += 32) {
      float4 v = xs4[i];
      float* p = ad + (i << 2);
      atomicAdd(&p[0], v.x * alpha);
      atomicAdd(&p[1], v.y * alpha);
      atomicAdd(&p[2], v.z * alpha);
      atomicAdd(&p[3], v.w * alpha);
    }
  } else {
    for (int c = lane; c < Cout; c += 32) atomicAdd(&ad[c], xs[c] * alpha);
  }
}

// ELU + fp16 copy for next layer's GEMM operand
__global__ void k_elu_cvt(float* h, _Float16* h16, const float* __restrict__ agg,
                          long n) {
  long t = (long)blockIdx.x * blockDim.x + threadIdx.x;
  if (t >= n) return;
  float v = agg[t];
  float o = v > 0.0f ? v : expm1f(v);
  h[t] = o;
  h16[t] = (_Float16)o;
}

// log_softmax over C=40 with broadcast bias xb[n]; wave per node
__global__ void k_final(float* out, const float* __restrict__ agg,
                        const float* __restrict__ xb, int N, int C) {
  int wid  = (blockIdx.x * blockDim.x + threadIdx.x) >> 5;
  int lane = threadIdx.x & 31;
  if (wid >= N) return;
  float b = xb[wid];
  const float* a = agg + (long)wid * C;
  float mx = -3.4e38f;
  for (int c = lane; c < C; c += 32) mx = fmaxf(mx, a[c] + b);
  mx = wave_max32(mx);
  float se = 0.0f;
  for (int c = lane; c < C; c += 32) se += expf(a[c] + b - mx);
  se = wave_sum32(se);
  float ls = logf(se) + mx;
  float* o = out + (long)wid * C;
  for (int c = lane; c < C; c += 32) o[c] = a[c] + b - ls;
}

// ------------------------------- host side ----------------------------------

extern "C" void kernel_launch(void* const* d_in, const int* in_sizes, int n_in,
                              void* d_out, int out_size, void* d_ws, size_t ws_size,
                              hipStream_t stream) {
  (void)in_sizes; (void)n_in; (void)out_size; (void)ws_size;
  const int N = N_NODES, E = N_EDGES;

  const float* x   = (const float*)d_in[0];
  const int*   ei  = (const int*)d_in[1];
  const int*   src = ei;
  const int*   dst = ei + E;
  const float* W0  = (const float*)d_in[2];
  const float* g0  = (const float*)d_in[3];
  const float* b0  = (const float*)d_in[4];
  const float* B0  = (const float*)d_in[5];
  const float* bb0 = (const float*)d_in[6];
  const float* W1  = (const float*)d_in[7];
  const float* g1  = (const float*)d_in[8];
  const float* b1  = (const float*)d_in[9];
  const float* B1  = (const float*)d_in[10];
  const float* bb1 = (const float*)d_in[11];
  const float* W2  = (const float*)d_in[12];
  const float* g2  = (const float*)d_in[13];
  const float* b2  = (const float*)d_in[14];
  const float* B2  = (const float*)d_in[15];
  const float* bb2 = (const float*)d_in[16];
  float* out = (float*)d_out;

  // workspace carve-up (256B aligned)
  char* wsb = (char*)d_ws;
  size_t off = 0;
  auto take = [&](size_t bytes) -> void* {
    void* p = wsb + off;
    off += (bytes + 255) & ~(size_t)255;
    return p;
  };
  float*    deg = (float*)take((size_t)N * 4);
  float*    nrm = (float*)take((size_t)N * 4);
  float*    xb  = (float*)take((size_t)N * 4);
  float*    mu  = (float*)take((size_t)N * 256 * 4);
  float*    xl  = (float*)take((size_t)N * 256 * 4);
  float*    agg = (float*)take((size_t)N * 256 * 4);
  float*    h   = (float*)take((size_t)N * 256 * 4);
  _Float16* h16 = (_Float16*)take((size_t)N * 256 * 2);
  _Float16* Wt  = (_Float16*)take((size_t)256 * 256 * 2);
  _Float16* Bt  = (_Float16*)take((size_t)256 * 256 * 2);

  auto GB = [](long threads) { return dim3((unsigned)((threads + 255) / 256)); };
  const dim3 TB(256);

  auto launch_gemm = [&](const _Float16* A, const _Float16* B16,
                         const float* bias, float* o,
                         int K, int Npad, int Nc, int ldo) {
    int ntiles = Npad >> 4;
    if ((ntiles & 3) == 0) {
      long waves = (long)(N / 16) * (ntiles / 4);
      k_gemm_wmma<4><<<GB(waves * 32), TB, 0, stream>>>(A, B16, bias, o,
                                                        N, K, Npad, Nc, ldo);
    } else if (ntiles % 3 == 0) {
      long waves = (long)(N / 16) * (ntiles / 3);
      k_gemm_wmma<3><<<GB(waves * 32), TB, 0, stream>>>(A, B16, bias, o,
                                                        N, K, Npad, Nc, ldo);
    } else {
      long waves = (long)(N / 16) * ntiles;
      k_gemm_wmma<1><<<GB(waves * 32), TB, 0, stream>>>(A, B16, bias, o,
                                                        N, K, Npad, Nc, ldo);
    }
  };

  // degrees (edge-structure only, computed once)
  k_zero<<<GB(N), TB, 0, stream>>>(deg, N);
  k_deg<<<GB(E), TB, 0, stream>>>(deg, src, E);
  // fp16 copy of layer-0 input for the GEMMs
  k_cvt_f16<<<GB((long)N * 128), TB, 0, stream>>>(h16, x, (long)N * 128);

  auto run_layer = [&](const float* xin, int Cin, int Cout, int Npad,
                       int H, int Ch, const float* W, const float* Bw,
                       const float* bb, const float* gam, const float* bet,
                       bool last) {
    // neighborhood mean + norms
    k_zero<<<GB((long)N * Cin), TB, 0, stream>>>(mu, (long)N * Cin);
    k_mu_acc<<<GB((long)E * (Cin / 4)), TB, 0, stream>>>(mu, xin, src, dst, E, Cin);
    k_mu_div<<<GB((long)N * Cin), TB, 0, stream>>>(mu, deg, N, Cin);
    k_norm<<<GB((long)N * 32), TB, 0, stream>>>(nrm, mu, N, Cin);

    // weight prep (transpose + fp16, padded columns)
    k_wtrans<<<GB((long)Npad * Cin), TB, 0, stream>>>(Wt, W, Cin, Cout, Npad);
    int bNpad = last ? 16 : Npad;
    int bNc   = last ? 1  : Cout;
    k_wtrans<<<GB((long)bNpad * Cin), TB, 0, stream>>>(Bt, Bw, Cin, bNc, bNpad);

    // xl = x @ W   (WMMA), then per-head LayerNorm in place
    launch_gemm(h16, Wt, nullptr, xl, Cin, Npad, Cout, Cout);
    k_ln<<<GB((long)N * H * 32), TB, 0, stream>>>(xl, gam, bet, N * H, Ch);

    // bias path
    if (!last) {
      launch_gemm(h16, Bt, bb, agg, Cin, Npad, Cout, Cout);
    } else {
      launch_gemm(h16, Bt, bb, xb, Cin, 16, 1, 1);
      k_zero<<<GB((long)N * Cout), TB, 0, stream>>>(agg, (long)N * Cout);
    }

    // self loops + edge messages (cosine attention, scatter-add)
    k_self<<<GB((long)N * Cout), TB, 0, stream>>>(agg, xl, nrm, N, Cout);
    k_edge<<<GB((long)E * 32), TB, 0, stream>>>(agg, xl, mu, nrm, src, dst,
                                                E, Cin, Cout);
    if (!last) {
      k_elu_cvt<<<GB((long)N * Cout), TB, 0, stream>>>(h, h16, agg,
                                                       (long)N * Cout);
    } else {
      k_final<<<GB((long)N * 32), TB, 0, stream>>>(out, agg, xb, N, Cout);
    }
  };

  run_layer(x, 128, 256, 256, 4, 64, W0, B0, bb0, g0, b0, false);
  run_layer(h, 256, 256, 256, 4, 64, W1, B1, bb1, g1, b1, false);
  run_layer(h, 256, 40, 48, 1, 40, W2, B2, bb2, g2, b2, true);
}
